// ConvAttention_41025527611451
// MI455X (gfx1250) — compile-verified, matrix operand order checked by reading, not run on hardware
//
#include <hip/hip_runtime.h>
#include <hip/hip_bf16.h>

// ---------------------------------------------------------------------------
// ConvAttention for MI455X (gfx1250, wave32, WMMA).
// All matrix work runs on v_wmma_f32_16x16x32_f16 (f16 in, f32 accumulate).
// Key idea this revision: choose memory layouts so that EVERY WMMA fragment
// run (8 consecutive K values per lane) is 8 contiguous f16 in memory:
//   * activations stored (B, T, C)  -- channel contiguous
//   * weights reordered (M, KTAP, Cin) -- K-order = (ktap major, channel minor)
// => A and B fragments are 2 x global_load_b128 each; conv outputs are one
// 16-byte packed-f16 store per row tile (D rows m = tile+v+8*hi are
// consecutive in the (B,T,M) layout). No per-element gathers remain.
// Zeros that matter (K-tail, conv padding) are whole-run value selects so
// EXEC stays all-ones across WMMA. Working set (~25 MB) is L2-resident.
// ---------------------------------------------------------------------------

typedef __attribute__((ext_vector_type(16))) _Float16 v16h;
typedef __attribute__((ext_vector_type(8)))  _Float16 v8h;
typedef __attribute__((ext_vector_type(8)))  float    v8f;
typedef __attribute__((ext_vector_type(4)))  float    v4f;

#define B_  8
#define CM  80
#define T1_ 800
#define CT  512
#define T2_ 200
#define CA  80

#define CAT16(a, b) __builtin_shufflevector(a, b, 0,1,2,3,4,5,6,7,8,9,10,11,12,13,14,15)

// ---------------------------------------------------------------------------
// Pre-pass: (B,C,T) f32 -> (B,T,C) f16   (contiguous f16 writes)
// ---------------------------------------------------------------------------
__global__ __launch_bounds__(256)
void transpose_cvt_kernel(const float* __restrict__ src, _Float16* __restrict__ dst,
                          int Bn, int C, int T)
{
    int i = blockIdx.x * blockDim.x + threadIdx.x;
    int total = Bn * C * T;
    if (i >= total) return;
    int c = i % C;
    int t = (i / C) % T;
    int b = i / (C * T);
    dst[i] = (_Float16)src[((size_t)b * C + c) * T + t];
}

// ---------------------------------------------------------------------------
// Pre-pass: weights (M, Cin, KTAP) f32 -> (M, KTAP, Cin) f16
// ---------------------------------------------------------------------------
__global__ __launch_bounds__(256)
void weight_cvt_kernel(const float* __restrict__ src, _Float16* __restrict__ dst,
                       int M, int Cin, int KTAP)
{
    int i = blockIdx.x * blockDim.x + threadIdx.x;
    int total = M * Cin * KTAP;
    if (i >= total) return;
    int c  = i % Cin;
    int kt = (i / Cin) % KTAP;
    int m  = i / (Cin * KTAP);
    dst[i] = (_Float16)src[((size_t)m * Cin + c) * KTAP + kt];
}

// ---------------------------------------------------------------------------
// Implicit-im2col conv1d as WMMA GEMM (all-f16 operands, f32 accumulate):
//   Y[b][t][m] = act( bias[m] + sum_{kt,c} W[m][kt][c] * X[b][t+kt-pad][c] )
// One wave per 16-wide t tile; MT row tiles reuse the B fragment.
// Requires: M % (16*MT) == 0, Cin % 8 == 0, Cin >= 32.
// ---------------------------------------------------------------------------
template <int MT, int KTAP, bool RELU>
__global__ __launch_bounds__(32)
void conv_gemm_wmma(const _Float16* __restrict__ X,  // (B, Tin, Cin) f16
                    const _Float16* __restrict__ W,  // (M, KTAP, Cin) f16
                    const float* __restrict__ bias,  // (M) f32
                    _Float16* __restrict__ Y,        // (B, Tout, M) f16
                    int M, int Cin, int Tin, int Tout, int pad)
{
    const int Keff    = Cin * KTAP;
    const int ntiles  = (Tout + 15) >> 4;
    const int mgroups = M / (16 * MT);       // exact by construction

    int gid = blockIdx.x;
    int nt  = gid % ntiles;  gid /= ntiles;
    int mg  = gid % mgroups;
    int b   = gid / mgroups;

    const int rc = threadIdx.x & 15;         // A-row / B-col inside tile
    const int hi = (threadIdx.x >> 4) & 1;

    const _Float16* Xb = X + (size_t)b * Tin * Cin;
    const int tcol   = nt * 16 + rc;
    const int tclamp = min(tcol, Tout - 1);  // OOB cols discarded at store

    const _Float16* Wrow[MT];
#pragma unroll
    for (int mt = 0; mt < MT; ++mt) {
        int mrow = (mg * MT + mt) * 16 + rc; // < M by construction
        Wrow[mt] = W + (size_t)mrow * Keff;
    }

    v8f acc[MT];
#pragma unroll
    for (int i = 0; i < MT; ++i) acc[i] = (v8f){0.f,0.f,0.f,0.f,0.f,0.f,0.f,0.f};

    // Fragment run state: run0 starts at K = 8*hi, run1 at K = 16+8*hi;
    // both advance by 32 per chunk. (ktap, c) tracked incrementally.
    int k0 = 8 * hi,      k1 = 16 + 8 * hi;
    int c0 = k0, kt0 = 0, c1 = k1, kt1 = 0;

    const int kchunks = (Keff + 31) >> 5;
    for (int kc = 0; kc < kchunks; ++kc) {
        // ---- B fragment: two contiguous 16-byte runs ---------------------
        v8h bz = {};
        v8h b0, b1;
        if (KTAP == 1) {
            int o0 = min(k0, Keff - 8), o1 = min(k1, Keff - 8);
            const _Float16* xr = Xb + (size_t)tclamp * Cin;
            v8h v0 = *(const v8h*)(xr + o0);
            v8h v1 = *(const v8h*)(xr + o1);
            b0 = (k0 < Keff) ? v0 : bz;
            b1 = (k1 < Keff) ? v1 : bz;
        } else {
            int ts0 = tclamp + kt0 - pad, ts1 = tclamp + kt1 - pad;
            bool ok0 = (kt0 < KTAP) & (ts0 >= 0) & (ts0 < Tin);
            bool ok1 = (kt1 < KTAP) & (ts1 >= 0) & (ts1 < Tin);
            int t0c = min(max(ts0, 0), Tin - 1);
            int t1c = min(max(ts1, 0), Tin - 1);
            v8h v0 = *(const v8h*)(Xb + (size_t)t0c * Cin + c0);
            v8h v1 = *(const v8h*)(Xb + (size_t)t1c * Cin + c1);
            b0 = ok0 ? v0 : bz;
            b1 = ok1 ? v1 : bz;
        }
        v16h bfrag = CAT16(b0, b1);

        // ---- A fragments (2 x b128 each, clamped; B zeros the K-tail) ----
        int a0 = min(k0, Keff - 8), a1 = min(k1, Keff - 8);
#pragma unroll
        for (int mt = 0; mt < MT; ++mt) {
            v8h w0 = *(const v8h*)(Wrow[mt] + a0);
            v8h w1 = *(const v8h*)(Wrow[mt] + a1);
            v16h afrag = CAT16(w0, w1);
            acc[mt] = __builtin_amdgcn_wmma_f32_16x16x32_f16(
                false, afrag, false, bfrag, (short)0, acc[mt], false, false);
        }

        // ---- advance run state (single wrap: Cin >= 32) ------------------
        k0 += 32; k1 += 32;
        if (KTAP > 1) {
            c0 += 32; if (c0 >= Cin) { c0 -= Cin; ++kt0; }
            c1 += 32; if (c1 >= Cin) { c1 -= Cin; ++kt1; }
        }
    }

    if (tcol < Tout) {
        _Float16* Yrow = Y + ((size_t)b * Tout + tcol) * M;
#pragma unroll
        for (int mt = 0; mt < MT; ++mt) {
            int mbase = (mg * MT + mt) * 16 + 8 * hi;  // D rows are contiguous
            v4f bs0 = *(const v4f*)(bias + mbase);
            v4f bs1 = *(const v4f*)(bias + mbase + 4);
            v8h hv;
#pragma unroll
            for (int v = 0; v < 8; ++v) {
                float r = acc[mt][v] + (v < 4 ? bs0[v] : bs1[v - 4]);
                if (RELU) r = fmaxf(r, 0.f);
                hv[v] = (_Float16)r;
            }
            *(v8h*)(Yrow + mbase) = hv;                // one 16B store
        }
    }
}

// ---------------------------------------------------------------------------
// Per-position squared norms of qe (B,T1,CA) and ke (B,T2,CA) — contiguous.
// ---------------------------------------------------------------------------
__global__ __launch_bounds__(256)
void norms_kernel(const _Float16* __restrict__ qe, const _Float16* __restrict__ ke,
                  float* __restrict__ qn, float* __restrict__ kn)
{
    int idx = blockIdx.x * blockDim.x + threadIdx.x;
    const int totQ = B_ * T1_;
    const _Float16* row;
    float* out;
    int oi;
    if (idx < totQ)            { row = qe + (size_t)idx * CA;            out = qn; oi = idx; }
    else if (idx < totQ + B_ * T2_) { int i2 = idx - totQ;
                                 row = ke + (size_t)i2 * CA;             out = kn; oi = i2;  }
    else return;
    float s = 0.f;
#pragma unroll
    for (int c8 = 0; c8 < CA; c8 += 8) {
        v8h v = *(const v8h*)(row + c8);
#pragma unroll
        for (int q = 0; q < 8; ++q) { float f = (float)v[q]; s += f * f; }
    }
    out[oi] = s;
}

// ---------------------------------------------------------------------------
// S[b][t1][t2] = -5e-4 * (|q|^2 + |k|^2) + 1e-3 * (q . k)  via WMMA GEMM
//   A[m=t1][k=c] = qe[b][t1][c],  B[k=c][n=t2] = ke[b][t2][c]; K=80 -> 96.
// All fragments are contiguous 16-byte runs; K loop fully unrolled.
// ---------------------------------------------------------------------------
__global__ __launch_bounds__(32)
void dist_gemm_wmma(const _Float16* __restrict__ qe, const _Float16* __restrict__ ke,
                    const float* __restrict__ qn, const float* __restrict__ kn,
                    float* __restrict__ S)
{
    const int ntiles = (T2_ + 15) >> 4;   // 13
    const int mtiles = T1_ >> 4;          // 50
    int gid = blockIdx.x;
    int nt  = gid % ntiles;  gid /= ntiles;
    int mt  = gid % mtiles;
    int b   = gid / mtiles;

    const int rc = threadIdx.x & 15;
    const int hi = (threadIdx.x >> 4) & 1;

    const int t1  = mt * 16 + rc;               // always < 800
    const int t2  = nt * 16 + rc;
    const int t2c = min(t2, T2_ - 1);           // OOB cols dropped at store

    const _Float16* Qr = qe + ((size_t)b * T1_ + t1)  * CA;
    const _Float16* Kr = ke + ((size_t)b * T2_ + t2c) * CA;

    v8f acc = (v8f){0.f,0.f,0.f,0.f,0.f,0.f,0.f,0.f};
#pragma unroll
    for (int kc = 0; kc < 3; ++kc) {
        int k0 = kc * 32 + 8 * hi, k1 = k0 + 16;
        int o0 = min(k0, CA - 8),  o1 = min(k1, CA - 8);
        v8h z = {};
        v8h qa0 = *(const v8h*)(Qr + o0);
        v8h qa1 = *(const v8h*)(Qr + o1);
        v8h kb0 = *(const v8h*)(Kr + o0);
        v8h kb1 = *(const v8h*)(Kr + o1);
        kb0 = (k0 < CA) ? kb0 : z;               // zero K-tail on B side only
        kb1 = (k1 < CA) ? kb1 : z;
        v16h av = CAT16(qa0, qa1);
        v16h bv = CAT16(kb0, kb1);
        acc = __builtin_amdgcn_wmma_f32_16x16x32_f16(
            false, av, false, bv, (short)0, acc, false, false);
    }

    if (t2 < T2_) {
#pragma unroll
        for (int v = 0; v < 8; ++v) {
            int m = mt * 16 + v + 8 * hi;
            S[((size_t)b * T1_ + m) * T2_ + t2] =
                1e-3f * acc[v] - 5e-4f * (qn[b * T1_ + m] + kn[b * T2_ + t2]);
        }
    }
}

// ---------------------------------------------------------------------------
// Row epilogue over t2 (one block per (b,t1)):
//   lp   = log_softmax(S) + log(prior + 1e-8)      -> out_logp
//   attn = softmax(mask ? -inf : lp)               -> out_attn
// ---------------------------------------------------------------------------
__global__ __launch_bounds__(256)
void softmax_kernel(const float* __restrict__ S, const float* __restrict__ prior,
                    const unsigned char* __restrict__ mask,
                    float* __restrict__ out_attn, float* __restrict__ out_logp)
{
    __shared__ float red[256];
    const int row = blockIdx.x;           // b*T1 + t1
    const int b   = row / T1_;
    const int t   = threadIdx.x;          // t2
    const bool valid = t < T2_;

    float s = valid ? S[(size_t)row * T2_ + t] : -INFINITY;

    red[t] = s; __syncthreads();
    for (int off = 128; off > 0; off >>= 1) {
        if (t < off) red[t] = fmaxf(red[t], red[t + off]);
        __syncthreads();
    }
    float mx = red[0]; __syncthreads();

    float e = valid ? __expf(s - mx) : 0.f;
    red[t] = e; __syncthreads();
    for (int off = 128; off > 0; off >>= 1) {
        if (t < off) red[t] += red[t + off];
        __syncthreads();
    }
    float lse = mx + __logf(red[0]); __syncthreads();

    float lp = valid ? (s - lse + __logf(prior[(size_t)row * T2_ + t] + 1e-8f))
                     : -INFINITY;
    bool msk = valid ? (mask[b * T2_ + t] != 0) : true;
    float x  = msk ? -INFINITY : lp;

    red[t] = x; __syncthreads();
    for (int off = 128; off > 0; off >>= 1) {
        if (t < off) red[t] = fmaxf(red[t], red[t + off]);
        __syncthreads();
    }
    float mx2 = red[0]; __syncthreads();
    float e2 = (x == -INFINITY) ? 0.f : __expf(x - mx2);
    red[t] = e2; __syncthreads();
    for (int off = 128; off > 0; off >>= 1) {
        if (t < off) red[t] += red[t + off];
        __syncthreads();
    }
    float denom = red[0];

    if (valid) {
        out_logp[(size_t)row * T2_ + t] = lp;
        out_attn[(size_t)row * T2_ + t] = e2 / denom;
    }
}

// ---------------------------------------------------------------------------
extern "C" void kernel_launch(void* const* d_in, const int* in_sizes, int n_in,
                              void* d_out, int out_size, void* d_ws, size_t ws_size,
                              hipStream_t stream)
{
    const float* queries = (const float*)d_in[0];   // (8,80,800)
    const float* keys    = (const float*)d_in[1];   // (8,512,200)
    // d_in[2] = query_lens (unused by reference math)
    const unsigned char* mask = (const unsigned char*)d_in[3];  // (8,200,1) bool
    const float* prior = (const float*)d_in[4];     // (8,800,200)
    const float* kp_w1 = (const float*)d_in[5];
    const float* kp_b1 = (const float*)d_in[6];
    const float* kp_w2 = (const float*)d_in[7];
    const float* kp_b2 = (const float*)d_in[8];
    const float* qp_w1 = (const float*)d_in[9];
    const float* qp_b1 = (const float*)d_in[10];
    const float* qp_w2 = (const float*)d_in[11];
    const float* qp_b2 = (const float*)d_in[12];
    const float* qp_w3 = (const float*)d_in[13];
    const float* qp_b3 = (const float*)d_in[14];

    // ---- workspace layout: f32 region first, then 16B-aligned f16 region --
    float* S  = (float*)d_ws;                        // 8*800*200
    float* qn = S  + (size_t)B_ * T1_ * T2_;         // 8*800
    float* kn = qn + (size_t)B_ * T1_;               // 8*200

    _Float16* h      = (_Float16*)(kn + (size_t)B_ * T2_);
    _Float16* keys_h = h;                      h += (size_t)B_ * T2_ * CT;   // (8,200,512)
    _Float16* qry_h  = h;                      h += (size_t)B_ * T1_ * CM;   // (8,800,80)
    _Float16* w1k_h  = h;                      h += (size_t)1024 * 3 * CT;   // (1024,3,512)
    _Float16* w2k_h  = h;                      h += (size_t)CA * 1024;       // (80,1,1024)
    _Float16* w1q_h  = h;                      h += (size_t)160 * 3 * CM;    // (160,3,80)
    _Float16* w2q_h  = h;                      h += (size_t)CM * 160;        // (80,1,160)
    _Float16* w3q_h  = h;                      h += (size_t)CA * CM;         // (80,1,80)
    _Float16* ke1_h  = h;                      h += (size_t)B_ * T2_ * 1024; // (8,200,1024)
    _Float16* qe1_h  = h;                      h += (size_t)B_ * T1_ * 160;  // (8,800,160)
    _Float16* ke_h   = h;                      h += (size_t)B_ * T2_ * CA;   // (8,200,80)
    _Float16* qe2_h  = h;                      h += (size_t)B_ * T1_ * CM;   // (8,800,80)
    _Float16* qe_h   = h;                      /* (8,800,80) */

    // ---- one-shot transpose/convert of inputs + weights -------------------
    transpose_cvt_kernel<<<(B_ * CT * T2_ + 255) / 256, 256, 0, stream>>>(
        keys, keys_h, B_, CT, T2_);
    transpose_cvt_kernel<<<(B_ * CM * T1_ + 255) / 256, 256, 0, stream>>>(
        queries, qry_h, B_, CM, T1_);
    weight_cvt_kernel<<<(1024 * CT * 3 + 255) / 256, 256, 0, stream>>>(
        kp_w1, w1k_h, 1024, CT, 3);
    weight_cvt_kernel<<<(CA * 1024 + 255) / 256, 256, 0, stream>>>(
        kp_w2, w2k_h, CA, 1024, 1);
    weight_cvt_kernel<<<(160 * CM * 3 + 255) / 256, 256, 0, stream>>>(
        qp_w1, w1q_h, 160, CM, 3);
    weight_cvt_kernel<<<(CM * 160 + 255) / 256, 256, 0, stream>>>(
        qp_w2, w2q_h, CM, 160, 1);
    weight_cvt_kernel<<<(CA * CM + 255) / 256, 256, 0, stream>>>(
        qp_w3, w3q_h, CA, CM, 1);

    dim3 blk(32);

    // key_proj: conv1d(512->1024,k3,p1)+ReLU ; conv1d(1024->80,k1)
    conv_gemm_wmma<4, 3, true ><<<13 * 16 * B_, blk, 0, stream>>>(
        keys_h, w1k_h, kp_b1, ke1_h, 1024, CT, T2_, T2_, 1);
    conv_gemm_wmma<5, 1, false><<<13 * 1 * B_, blk, 0, stream>>>(
        ke1_h, w2k_h, kp_b2, ke_h, CA, 1024, T2_, T2_, 0);

    // query_proj: conv(80->160,k3,p1)+ReLU ; conv(160->80,k1)+ReLU ; conv(80->80,k1)
    conv_gemm_wmma<5, 3, true ><<<50 * 2 * B_, blk, 0, stream>>>(
        qry_h, w1q_h, qp_b1, qe1_h, 160, CM, T1_, T1_, 1);
    conv_gemm_wmma<5, 1, true ><<<50 * 1 * B_, blk, 0, stream>>>(
        qe1_h, w2q_h, qp_b2, qe2_h, CM, 160, T1_, T1_, 0);
    conv_gemm_wmma<5, 1, false><<<50 * 1 * B_, blk, 0, stream>>>(
        qe2_h, w3q_h, qp_b3, qe_h, CA, CM, T1_, T1_, 0);

    // norms + distance GEMM + softmax epilogue
    norms_kernel<<<(B_ * (T1_ + T2_) + 255) / 256, 256, 0, stream>>>(
        qe_h, ke_h, qn, kn);
    dist_gemm_wmma<<<13 * 50 * B_, blk, 0, stream>>>(qe_h, ke_h, qn, kn, S);

    float* out_attn = (float*)d_out;                       // (8,1,800,200)
    float* out_logp = out_attn + (size_t)B_ * T1_ * T2_;   // (8,1,800,200)
    softmax_kernel<<<B_ * T1_, 256, 0, stream>>>(S, prior, mask, out_attn, out_logp);
}
